// MultiScaleARSPBlock_45535243272924
// MI455X (gfx1250) — compile-verified, matrix operand order checked by reading, not run on hardware
//
#include <hip/hip_runtime.h>
#include <hip/hip_bf16.h>

// ---------------------------------------------------------------------------
// MI455X (gfx1250) implementation.
//  - wave32; WMMA f32_16x16x32_f16 for all big GEMMs (fp32 accumulate)
//  - GLOBAL_LOAD_ASYNC_TO_LDS_B128 + s_wait_asynccnt double-buffered A staging
//  - weights pre-packed once per launch into fragment-order packed-f16 words:
//    one dword load per B fragment slot, fully coalesced, no cvt in hot loop
//  - edge scatter: global_atomic_add_f32, LDS-staged edge weights
//  - fused head-GEMMs + MLP + 2x LayerNorm: no "cat" buffer, xs tiles in LDS
//  - SSM emits only the last token (only mo[:,-1] is consumed)
// ---------------------------------------------------------------------------

typedef __attribute__((ext_vector_type(16))) _Float16 v16h;
typedef __attribute__((ext_vector_type(2)))  _Float16 v2h;
typedef __attribute__((ext_vector_type(8)))  float    v8f;

union Frag {                 // WMMA A/B fragment: 8 VGPRs = 16 f16 = 8 dwords
  v16h v;
  v2h p[8];
  unsigned u[8];
};

#define Nn 50000
#define Ee 640000
#define Cc 128
#define Bb 64
#define Ll 32
#define Dd 256
#define Ss 128

__device__ __forceinline__ float sigmf(float v) { return 1.f / (1.f + expf(-v)); }

// low 32 bits of a generic pointer into LDS == wave-relative LDS byte offset
__device__ __forceinline__ unsigned lds_off(const void* p) {
  return (unsigned)(unsigned long long)p;
}

#define ASYNC_B128(lds, gp) \
  asm volatile("global_load_async_to_lds_b128 %0, %1, off" :: "v"(lds), "v"(gp) : "memory")
#define ASYNC_B128_OFF(lds, gp, imm) \
  asm volatile("global_load_async_to_lds_b128 %0, %1, off offset:" #imm :: "v"(lds), "v"(gp) : "memory")
#define WAIT_ASYNC(imm) \
  asm volatile("s_wait_asynccnt " #imm ::: "memory")

// ---------------------------------------------------------------- zero scratch
__global__ void zero_kernel(float* __restrict__ p, unsigned long long n) {
  unsigned long long i = (unsigned long long)blockIdx.x * blockDim.x + threadIdx.x;
  unsigned long long stride = (unsigned long long)gridDim.x * blockDim.x;
  for (; i < n; i += stride) p[i] = 0.f;
}

// ----------------------------------------- pack fp32 weights -> fragment f16x2
// P[k2*N + n] = {f16(B[2k2][n]), f16(B[2k2+1][n])}
__global__ void pack_f16(const float* __restrict__ B, unsigned* __restrict__ P,
                         int K2, int N) {
  const int idx = blockIdx.x * 256 + threadIdx.x;
  if (idx >= K2 * N) return;
  const int k2 = idx / N, n = idx - k2 * N;
  union { v2h p; unsigned u; } w;
  w.p = v2h{(_Float16)B[(unsigned long long)(2 * k2) * N + n],
            (_Float16)B[(unsigned long long)(2 * k2 + 1) * N + n]};
  P[idx] = w.u;
}

// ------------------------------------------------------- edge messages+scatter
__global__ __launch_bounds__(256) void edge_scatter(
    const float* __restrict__ x, const float* __restrict__ eattr,
    const int* __restrict__ srci, const int* __restrict__ dsti,
    const float* __restrict__ We, const float* __restrict__ be,
    float* __restrict__ agg)
{
  __shared__ float sWe[8 * 2 * Cc];
  __shared__ float sbe[8 * Cc];
  for (int i = threadIdx.x; i < 8 * 2 * Cc; i += 256) sWe[i] = We[i];
  for (int i = threadIdx.x; i < 8 * Cc; i += 256) sbe[i] = be[i];
  __syncthreads();
  const int wv = threadIdx.x >> 5, lane = threadIdx.x & 31;
  const int e = blockIdx.x * 8 + wv;
  if (e >= Ee) return;
  const int s = srci[e], d = dsti[e];
  const float a0 = eattr[2 * e], a1 = eattr[2 * e + 1];
  const int c = lane * 4;
  const float4 x4 = *reinterpret_cast<const float4*>(x + (unsigned long long)s * Cc + c);
  const float xa[4] = {x4.x, x4.y, x4.z, x4.w};
  #pragma unroll
  for (int h = 0; h < 8; ++h) {
    const float* w0 = &sWe[(h * 2 + 0) * Cc + c];
    const float* w1 = &sWe[(h * 2 + 1) * Cc + c];
    const float* bb = &sbe[h * Cc + c];
    float* dp = agg + ((unsigned long long)h * Nn + d) * Cc + c;
    #pragma unroll
    for (int i = 0; i < 4; ++i) {
      float m = xa[i] + a0 * w0[i] + a1 * w1[i] + bb[i];
      m = fmaxf(m, 0.f);
      atomicAdd(dp + i, m);
    }
  }
}

// ----------------------------------------------------------------- batch pool
__global__ void pool_sum(const float* __restrict__ x, const int* __restrict__ batch,
                         float* __restrict__ xsum, int* __restrict__ counts) {
  const int idx = blockIdx.x * 256 + threadIdx.x;
  if (idx >= Nn * Cc) return;
  const int nrow = idx >> 7, c = idx & 127;
  const int b = batch[nrow];
  atomicAdd(&xsum[b * Cc + c], x[idx]);
  if (c == 0) atomicAdd(&counts[b], 1);
}

__global__ void pool_fin(const float* __restrict__ xsum, const int* __restrict__ counts,
                         float* __restrict__ xmean) {
  __shared__ float maxN;
  if (threadIdx.x == 0) {
    int m = 0;
    for (int i = 0; i < Bb; ++i) m = max(m, counts[i]);
    maxN = (float)m;
  }
  __syncthreads();
  for (int i = threadIdx.x; i < Bb * Cc; i += 128) xmean[i] = xsum[i] / maxN;
}

// ----------------------------------------------------------------------- GRU
__global__ __launch_bounds__(128) void gru_kernel(
    const float* __restrict__ xmean, const float* __restrict__ Wih,
    const float* __restrict__ Whh, const float* __restrict__ bih,
    const float* __restrict__ bhh, float* __restrict__ tokens)
{
  __shared__ float xm[Cc];
  __shared__ float gi[3 * Cc];
  __shared__ float hs[Cc];
  const int b = blockIdx.x, c = threadIdx.x;
  xm[c] = xmean[b * Cc + c];
  hs[c] = 0.f;
  __syncthreads();
  for (int j = c; j < 3 * Cc; j += Cc) {
    float acc = bih[j];
    for (int k = 0; k < Cc; ++k) acc += xm[k] * Wih[k * 3 * Cc + j];
    gi[j] = acc;
  }
  __syncthreads();
  for (int t = 0; t < Ll; ++t) {
    float ghr = bhh[c], ghz = bhh[Cc + c], ghn = bhh[2 * Cc + c];
    for (int k = 0; k < Cc; ++k) {
      const float hk = hs[k];
      ghr += hk * Whh[k * 3 * Cc + c];
      ghz += hk * Whh[k * 3 * Cc + Cc + c];
      ghn += hk * Whh[k * 3 * Cc + 2 * Cc + c];
    }
    const float r = sigmf(gi[c] + ghr);
    const float z = sigmf(gi[Cc + c] + ghz);
    const float n = tanhf(gi[2 * Cc + c] + r * ghn);
    const float hnew = (1.f - z) * n + z * hs[c];
    __syncthreads();
    hs[c] = hnew;
    tokens[((unsigned long long)b * Ll + t) * Cc + c] = hnew;
    __syncthreads();
  }
}

// -------------------------------------------------------- generic WMMA GEMM
// C[M,Ncols] = A[M,K] @ B[K,Ncols]; A fp32 (async-staged in LDS, double
// buffered), B pre-packed f16x2 in fragment order, fp32 accumulate.
__global__ __launch_bounds__(128) void gemm_wmma(
    const float* __restrict__ A, int lda, const unsigned* __restrict__ Bp,
    float* __restrict__ C, int ldc, int K, int Ncols)
{
  __shared__ float As[2][16 * 32];
  const int tid = threadIdx.x;
  const int lane = tid & 31, wv = tid >> 5;
  const int row0 = blockIdx.x * 16;
  const int ct = blockIdx.y * 4 + wv;
  const int col0 = ct * 16;
  const int mA = lane & 15, half = lane >> 4;
  const int n = col0 + mA;
  const bool nok = n < Ncols;
  const int nc = nok ? n : (Ncols - 1);      // clamped, in-bounds address
  const int sr = tid >> 3;                   // staging row 0..15
  const int sc = (tid & 7) * 4;              // staging col 0,4,...,28

  v8f acc = {};
  {
    const float* gp = A + (unsigned long long)(row0 + sr) * lda + sc;
    ASYNC_B128(lds_off(&As[0][sr * 32 + sc]), gp);
  }
  const int nk = K / 32;
  for (int ik = 0; ik < nk; ++ik) {
    if (ik + 1 < nk) {
      const float* gp = A + (unsigned long long)(row0 + sr) * lda + (ik + 1) * 32 + sc;
      ASYNC_B128(lds_off(&As[(ik + 1) & 1][sr * 32 + sc]), gp);
      WAIT_ASYNC(0x1);
    } else {
      WAIT_ASYNC(0x0);
    }
    __syncthreads();
    const float* Ab = As[ik & 1];
    Frag af, bf;
    #pragma unroll
    for (int v = 0; v < 8; ++v) {
      const int k = (v & 3) * 2 + ((v >> 2) << 4) + (half << 3);
      const int k2 = (v & 3) + ((v >> 2) << 3) + (half << 2);
      af.p[v] = v2h{(_Float16)Ab[mA * 32 + k], (_Float16)Ab[mA * 32 + k + 1]};
      const unsigned w = Bp[(unsigned long long)(ik * 16 + k2) * Ncols + nc];
      bf.u[v] = nok ? w : 0u;
    }
    acc = __builtin_amdgcn_wmma_f32_16x16x32_f16(false, af.v, false, bf.v, (short)0, acc, false, false);
    __syncthreads();
  }
  #pragma unroll
  for (int v = 0; v < 8; ++v) {
    const int m = v + 8 * half;
    if (nok) C[(unsigned long long)(row0 + m) * ldc + n] = acc[v];
  }
}

// --------------------------------------------------------------- conv + silu
__global__ void conv_silu(const float* __restrict__ xz, const float* __restrict__ cw,
                          const float* __restrict__ cb, float* __restrict__ x_c) {
  const int idx = blockIdx.x * 256 + threadIdx.x;
  if (idx >= Bb * Ll * Dd) return;
  const int d = idx & (Dd - 1);
  const int bt = idx >> 8;
  const int t = bt & (Ll - 1);
  const int b = bt >> 5;
  float acc = cb[d];
  #pragma unroll
  for (int k = 0; k < 4; ++k) {
    const int tt = t + k - 3;
    if (tt >= 0) acc += cw[d * 4 + k] * xz[((unsigned long long)(b * Ll + tt)) * (2 * Dd) + d];
  }
  x_c[idx] = acc * sigmf(acc);
}

// --------------------------------------------------------------------- dt
__global__ void dt_kernel(const float* __restrict__ dbl, const float* __restrict__ Wdt,
                          const float* __restrict__ bdt, float* __restrict__ dtb) {
  const int idx = blockIdx.x * 256 + threadIdx.x;
  if (idx >= Bb * Ll * Dd) return;
  const int d = idx & (Dd - 1);
  const int bt = idx >> 8;
  float acc = bdt[d];
  #pragma unroll
  for (int r = 0; r < 8; ++r) acc += dbl[(unsigned long long)bt * 264 + r] * Wdt[r * Dd + d];
  dtb[idx] = (acc > 20.f) ? acc : log1pf(expf(acc));
}

// -------------------------------------------------------------- SSM scan
__global__ __launch_bounds__(256) void ssm_scan(
    const float* __restrict__ dtb, const float* __restrict__ dbl,
    const float* __restrict__ x_c, const float* __restrict__ xz,
    const float* __restrict__ A_log, const float* __restrict__ Dp,
    float* __restrict__ ylast)
{
  const int b = blockIdx.x;
  const int dg = blockIdx.y;
  const int lane = threadIdx.x & 31;
  const int dl = threadIdx.x >> 5;
  const int d = dg * 8 + dl;
  const int s0 = lane * 4;
  float a[4], h[4] = {0.f, 0.f, 0.f, 0.f};
  #pragma unroll
  for (int i = 0; i < 4; ++i) a[i] = -expf(A_log[d * Ss + s0 + i]);
  for (int t = 0; t < Ll; ++t) {
    const unsigned long long bt = (unsigned long long)b * Ll + t;
    const float dtv = dtb[bt * Dd + d];
    const float xcv = x_c[bt * Dd + d];
    const float4 Bp = *reinterpret_cast<const float4*>(dbl + bt * 264 + 8 + s0);
    const float bv[4] = {Bp.x, Bp.y, Bp.z, Bp.w};
    const float dx = dtv * xcv;
    #pragma unroll
    for (int i = 0; i < 4; ++i) h[i] = expf(dtv * a[i]) * h[i] + dx * bv[i];
    if (t == Ll - 1) {
      const float4 Cp = *reinterpret_cast<const float4*>(dbl + bt * 264 + 136 + s0);
      float part = h[0] * Cp.x + h[1] * Cp.y + h[2] * Cp.z + h[3] * Cp.w;
      for (int off = 16; off > 0; off >>= 1) part += __shfl_xor(part, off, 32);
      if (lane == 0) {
        const float zg = xz[bt * (2 * Dd) + Dd + d];
        ylast[b * Dd + d] = (part + Dp[d] * xcv) * (zg * sigmf(zg));
      }
    }
  }
}

// ------------------------------------------------------------- row LayerNorm
__global__ __launch_bounds__(128) void ln_rows(const float* __restrict__ in,
                                               const float* __restrict__ g,
                                               const float* __restrict__ bb,
                                               float* __restrict__ out) {
  const int r = blockIdx.x, c = threadIdx.x;
  __shared__ float ssum[Cc], ssq[Cc];
  const float v = in[r * Cc + c];
  ssum[c] = v; ssq[c] = v * v;
  __syncthreads();
  for (int off = 64; off > 0; off >>= 1) {
    if (c < off) { ssum[c] += ssum[c + off]; ssq[c] += ssq[c + off]; }
    __syncthreads();
  }
  const float mean = ssum[0] * (1.f / Cc);
  const float var = ssq[0] * (1.f / Cc) - mean * mean;
  out[r * Cc + c] = (v - mean) * rsqrtf(var + 1e-5f) * g[c] + bb[c];
}

// ---------------------------------------------------- fused heads + MLP + LN
__global__ __launch_bounds__(128) void fused_mlp(
    const float* __restrict__ agg, const unsigned* __restrict__ Wcp,
    const float* __restrict__ bc,
    const unsigned* __restrict__ Wm1p, const float* __restrict__ bm1,
    const float* __restrict__ g_h, const float* __restrict__ b_h,
    const unsigned* __restrict__ Wm2p, const float* __restrict__ bm2,
    const float* __restrict__ g_o, const float* __restrict__ b_o,
    const float* __restrict__ x, const float* __restrict__ x_m,
    const int* __restrict__ batch, float* __restrict__ out)
{
  __shared__ float    aggS[2][16 * Cc];   // async-staged agg tiles (fp32)
  __shared__ _Float16 segA[16 * Cc];
  __shared__ float    fbuf[16 * Cc];
  __shared__ _Float16 h1h[16 * Cc];
  __shared__ float redS[128], redQ[128];

  const int tid = threadIdx.x;
  const int lane = tid & 31, wv = tid >> 5;
  const int row0 = blockIdx.x * 16;
  const int mA = lane & 15, half = lane >> 4;
  const int sr = tid >> 3;                // staging row 0..15
  const int sc = (tid & 7) * 16;          // staging col 0,16,...,112

  v8f acc1[2] = {};

  // prefetch head 0 agg tile
  {
    const float* gp = agg + ((unsigned long long)0 * Nn + row0) * Cc + sr * Cc + sc;
    const unsigned lo = lds_off(&aggS[0][sr * Cc + sc]);
    ASYNC_B128_OFF(lo, gp, 0);
    ASYNC_B128_OFF(lo, gp, 16);
    ASYNC_B128_OFF(lo, gp, 32);
    ASYNC_B128_OFF(lo, gp, 48);
  }

  for (int seg = 0; seg < 9; ++seg) {
    if (seg < 8) {
      WAIT_ASYNC(0x0);          // our part of aggS[seg&1] done
      __syncthreads();          // everyone's part done; segA free (prev GEMM1 done)
      const float* Ab = aggS[seg & 1];
      const unsigned* Bh = Wcp + seg * (Cc / 2) * Cc;
      #pragma unroll
      for (int ct = 0; ct < 2; ++ct) {
        const int col0 = (wv * 2 + ct) * 16;
        v8f acc = {};
        for (int kk = 0; kk < Cc; kk += 32) {
          Frag af, bf;
          #pragma unroll
          for (int v = 0; v < 8; ++v) {
            const int k = kk + (v & 3) * 2 + ((v >> 2) << 4) + (half << 3);
            const int k2 = kk / 2 + (v & 3) + ((v >> 2) << 3) + (half << 2);
            af.p[v] = v2h{(_Float16)Ab[mA * Cc + k], (_Float16)Ab[mA * Cc + k + 1]};
            bf.u[v] = Bh[k2 * Cc + col0 + mA];
          }
          acc = __builtin_amdgcn_wmma_f32_16x16x32_f16(false, af.v, false, bf.v, (short)0, acc, false, false);
        }
        #pragma unroll
        for (int v = 0; v < 8; ++v) {
          const int m = v + 8 * half;
          const int n = col0 + mA;
          segA[m * Cc + n] = (_Float16)fmaxf(acc[v] + bc[seg * Cc + n], 0.f);
        }
      }
      __syncthreads();          // segA ready; aggS[(seg+1)&1] free (consumed at seg-1)
      if (seg + 1 < 8) {        // prefetch next head's agg tile (overlaps GEMM1)
        const float* gp = agg + ((unsigned long long)(seg + 1) * Nn + row0) * Cc + sr * Cc + sc;
        const unsigned lo = lds_off(&aggS[(seg + 1) & 1][sr * Cc + sc]);
        ASYNC_B128_OFF(lo, gp, 0);
        ASYNC_B128_OFF(lo, gp, 16);
        ASYNC_B128_OFF(lo, gp, 32);
        ASYNC_B128_OFF(lo, gp, 48);
      }
    } else {
      __syncthreads();          // segA free
      const int bidx = batch[row0 + sr];
      for (int i = 0; i < 16; ++i)
        segA[sr * Cc + sc + i] = (_Float16)x_m[bidx * Cc + sc + i];
      __syncthreads();          // segA ready
    }
    // GEMM1 partial: acc1 += segA @ Wm1[seg*128 .. seg*128+127]
    #pragma unroll
    for (int ct = 0; ct < 2; ++ct) {
      const int col0 = (wv * 2 + ct) * 16;
      for (int kk = 0; kk < Cc; kk += 32) {
        Frag af, bf;
        #pragma unroll
        for (int v = 0; v < 8; ++v) {
          const int k = kk + (v & 3) * 2 + ((v >> 2) << 4) + (half << 3);
          const int k2 = kk / 2 + (v & 3) + ((v >> 2) << 3) + (half << 2);
          af.p[v] = *reinterpret_cast<const v2h*>(&segA[mA * Cc + k]);
          bf.u[v] = Wm1p[(unsigned long long)(seg * (Cc / 2) + k2) * Cc + col0 + mA];
        }
        acc1[ct] = __builtin_amdgcn_wmma_f32_16x16x32_f16(false, af.v, false, bf.v, (short)0, acc1[ct], false, false);
      }
    }
    __syncthreads();
  }

  #pragma unroll
  for (int ct = 0; ct < 2; ++ct) {
    const int col0 = (wv * 2 + ct) * 16;
    const int n = col0 + mA;
    #pragma unroll
    for (int v = 0; v < 8; ++v) fbuf[(v + 8 * half) * Cc + n] = acc1[ct][v] + bm1[n];
  }
  __syncthreads();
  {
    const int r = tid >> 3, sg = tid & 7;
    float s = 0.f, q = 0.f;
    for (int i = 0; i < 16; ++i) { const float v = fbuf[r * Cc + sg * 16 + i]; s += v; q += v * v; }
    redS[tid] = s; redQ[tid] = q;
    __syncthreads();
    float ts = 0.f, tq = 0.f;
    for (int i = 0; i < 8; ++i) { ts += redS[r * 8 + i]; tq += redQ[r * 8 + i]; }
    const float mean = ts * (1.f / Cc);
    const float inv = rsqrtf(tq * (1.f / Cc) - mean * mean + 1e-5f);
    for (int i = 0; i < 16; ++i) {
      const int c = sg * 16 + i;
      const float v = (fbuf[r * Cc + c] - mean) * inv * g_h[c] + b_h[c];
      h1h[r * Cc + c] = (_Float16)fmaxf(v, 0.f);
    }
  }
  __syncthreads();
  v8f acc2[2] = {};
  #pragma unroll
  for (int ct = 0; ct < 2; ++ct) {
    const int col0 = (wv * 2 + ct) * 16;
    for (int kk = 0; kk < Cc; kk += 32) {
      Frag af, bf;
      #pragma unroll
      for (int v = 0; v < 8; ++v) {
        const int k = kk + (v & 3) * 2 + ((v >> 2) << 4) + (half << 3);
        const int k2 = kk / 2 + (v & 3) + ((v >> 2) << 3) + (half << 2);
        af.p[v] = *reinterpret_cast<const v2h*>(&h1h[mA * Cc + k]);
        bf.u[v] = Wm2p[k2 * Cc + col0 + mA];
      }
      acc2[ct] = __builtin_amdgcn_wmma_f32_16x16x32_f16(false, af.v, false, bf.v, (short)0, acc2[ct], false, false);
    }
  }
  __syncthreads();
  #pragma unroll
  for (int ct = 0; ct < 2; ++ct) {
    const int col0 = (wv * 2 + ct) * 16;
    const int n = col0 + mA;
    #pragma unroll
    for (int v = 0; v < 8; ++v) {
      const int m = v + 8 * half;
      fbuf[m * Cc + n] = acc2[ct][v] + bm2[n] + x[(unsigned long long)(row0 + m) * Cc + n];
    }
  }
  __syncthreads();
  {
    const int r = tid >> 3, sg = tid & 7;
    float s = 0.f, q = 0.f;
    for (int i = 0; i < 16; ++i) { const float v = fbuf[r * Cc + sg * 16 + i]; s += v; q += v * v; }
    redS[tid] = s; redQ[tid] = q;
    __syncthreads();
    float ts = 0.f, tq = 0.f;
    for (int i = 0; i < 8; ++i) { ts += redS[r * 8 + i]; tq += redQ[r * 8 + i]; }
    const float mean = ts * (1.f / Cc);
    const float inv = rsqrtf(tq * (1.f / Cc) - mean * mean + 1e-5f);
    for (int i = 0; i < 16; ++i) {
      const int c = sg * 16 + i;
      out[(unsigned long long)(row0 + r) * Cc + c] = (fbuf[r * Cc + c] - mean) * inv * g_o[c] + b_o[c];
    }
  }
}

// ---------------------------------------------------------------------------
extern "C" void kernel_launch(void* const* d_in, const int* in_sizes, int n_in,
                              void* d_out, int out_size, void* d_ws, size_t ws_size,
                              hipStream_t stream) {
  const float* x      = (const float*)d_in[0];
  const float* eattr  = (const float*)d_in[1];
  const int*   eidx   = (const int*)d_in[2];
  const int*   batch  = (const int*)d_in[3];
  const float* We     = (const float*)d_in[4];
  const float* be     = (const float*)d_in[5];
  const float* Wc     = (const float*)d_in[6];
  const float* bc     = (const float*)d_in[7];
  const float* Wih    = (const float*)d_in[8];
  const float* Whh    = (const float*)d_in[9];
  const float* bih    = (const float*)d_in[10];
  const float* bhh    = (const float*)d_in[11];
  const float* Win    = (const float*)d_in[12];
  const float* conv_w = (const float*)d_in[13];
  const float* conv_b = (const float*)d_in[14];
  const float* Wx     = (const float*)d_in[15];
  const float* Wdt    = (const float*)d_in[16];
  const float* bdt    = (const float*)d_in[17];
  const float* A_log  = (const float*)d_in[18];
  const float* Dp     = (const float*)d_in[19];
  const float* Wout   = (const float*)d_in[20];
  const float* g_m    = (const float*)d_in[21];
  const float* b_m    = (const float*)d_in[22];
  const float* Wm1    = (const float*)d_in[23];
  const float* bm1    = (const float*)d_in[24];
  const float* g_h    = (const float*)d_in[25];
  const float* b_h    = (const float*)d_in[26];
  const float* Wm2    = (const float*)d_in[27];
  const float* bm2    = (const float*)d_in[28];
  const float* g_o    = (const float*)d_in[29];
  const float* b_o    = (const float*)d_in[30];
  float* out = (float*)d_out;

  float* ws     = (float*)d_ws;
  float* agg    = ws;                                   // 8*N*C = 51,200,000
  float* xsum   = agg + (unsigned long long)8 * Nn * Cc;
  int*   counts = (int*)(xsum + Bb * Cc);
  float* xmean  = (float*)(counts + Bb);
  float* tokens = xmean + Bb * Cc;                      // B*L*C
  float* xz     = tokens + Bb * Ll * Cc;                // B*L*2D
  float* x_c    = xz + (unsigned long long)Bb * Ll * 2 * Dd;
  float* dbl    = x_c + (unsigned long long)Bb * Ll * Dd;     // B*L*264
  float* dtb    = dbl + (unsigned long long)Bb * Ll * 264;
  float* ylast  = dtb + (unsigned long long)Bb * Ll * Dd;     // B*D
  float* mopre  = ylast + Bb * Dd;                            // B*C
  float* x_m    = mopre + Bb * Cc;                            // B*C
  // packed-f16 weights (u32 words, fragment order)
  unsigned* Wcp   = (unsigned*)(x_m + Bb * Cc);               // 512*128
  unsigned* Wm1p  = Wcp + 512 * 128;                          // 576*128
  unsigned* Wm2p  = Wm1p + 576 * 128;                         // 64*128
  unsigned* Winp  = Wm2p + 64 * 128;                          // 64*512
  unsigned* Wxp   = Winp + 64 * 512;                          // 128*264
  unsigned* Woutp = Wxp + 128 * 264;                          // 128*128

  const unsigned long long nzero = (unsigned long long)8 * Nn * Cc + Bb * Cc + Bb;
  zero_kernel<<<16384, 256, 0, stream>>>(ws, nzero);

  // weight repacks (tiny, run up front)
  pack_f16<<<(512 * 128 + 255) / 256, 256, 0, stream>>>(Wc, Wcp, 512, 128);
  pack_f16<<<(576 * 128 + 255) / 256, 256, 0, stream>>>(Wm1, Wm1p, 576, 128);
  pack_f16<<<(64 * 128 + 255) / 256, 256, 0, stream>>>(Wm2, Wm2p, 64, 128);
  pack_f16<<<(64 * 512 + 255) / 256, 256, 0, stream>>>(Win, Winp, 64, 512);
  pack_f16<<<(128 * 264 + 255) / 256, 256, 0, stream>>>(Wx, Wxp, 128, 264);
  pack_f16<<<(128 * 128 + 255) / 256, 256, 0, stream>>>(Wout, Woutp, 128, 128);

  edge_scatter<<<(Ee + 7) / 8, 256, 0, stream>>>(x, eattr, eidx, eidx + Ee, We, be, agg);

  pool_sum<<<(Nn * Cc + 255) / 256, 256, 0, stream>>>(x, batch, xsum, counts);
  pool_fin<<<1, 128, 0, stream>>>(xsum, counts, xmean);

  gru_kernel<<<Bb, 128, 0, stream>>>(xmean, Wih, Whh, bih, bhh, tokens);

  // xz = tokens @ Win   (2048 x 128) @ (128 x 512)
  gemm_wmma<<<dim3(Bb * Ll / 16, 8), 128, 0, stream>>>(tokens, Cc, Winp, xz, 2 * Dd, Cc, 2 * Dd);

  conv_silu<<<(Bb * Ll * Dd + 255) / 256, 256, 0, stream>>>(xz, conv_w, conv_b, x_c);

  // dbl = x_c @ Wx      (2048 x 256) @ (256 x 264)
  gemm_wmma<<<dim3(Bb * Ll / 16, 5), 128, 0, stream>>>(x_c, Dd, Wxp, dbl, 264, Dd, 264);

  dt_kernel<<<(Bb * Ll * Dd + 255) / 256, 256, 0, stream>>>(dbl, Wdt, bdt, dtb);

  ssm_scan<<<dim3(Bb, Dd / 8), 256, 0, stream>>>(dtb, dbl, x_c, xz, A_log, Dp, ylast);

  // mo_pre = y_last @ Wout   (64 x 256) @ (256 x 128)
  gemm_wmma<<<dim3(Bb / 16, 2), 128, 0, stream>>>(ylast, Dd, Woutp, mopre, Cc, Dd, Cc);
  ln_rows<<<Bb, 128, 0, stream>>>(mopre, g_m, b_m, x_m);

  fused_mlp<<<Nn / 16, 128, 0, stream>>>(agg, Wcp, bc, Wm1p, bm1, g_h, b_h,
                                         Wm2p, bm2, g_o, b_o, x, x_m, batch, out);

  (void)in_sizes; (void)n_in; (void)out_size; (void)ws_size;
}
